// Attention_48825188221010
// MI455X (gfx1250) — matrix operand order repaired and checked by reading, not compiled, not run on hardware
//
#include <hip/hip_runtime.h>

// ---------------------------------------------------------------------------
// Problem constants (from the reference)
// ---------------------------------------------------------------------------
#define SEQ     4096
#define DMODEL  2048
#define NHEADS  16
#define NKVH    4
#define HEADDIM 128
#define KVDIM   (NKVH * HEADDIM)   // 512

typedef __bf16        bfvec16 __attribute__((ext_vector_type(16)));
typedef float         v8f_t   __attribute__((ext_vector_type(8)));
typedef unsigned int  u32x4   __attribute__((ext_vector_type(4)));
typedef int           i32x4   __attribute__((ext_vector_type(4)));

#define AS1 __attribute__((address_space(1)))
#define AS3 __attribute__((address_space(3)))

#ifndef __has_builtin
#define __has_builtin(x) 0
#endif
#if __has_builtin(__builtin_amdgcn_global_load_async_to_lds_b128) && \
    __has_builtin(__builtin_amdgcn_s_wait_asynccnt)
#define USE_ASYNC_LDS 1
#else
#define USE_ASYNC_LDS 0
#endif

union FragBf {
    bfvec16 v;
    u32x4   u[2];
};

#if USE_ASYNC_LDS
static __device__ inline AS1 i32x4* as_global_v4(const __bf16* p) {
    return (AS1 i32x4*)p;
}
static __device__ inline AS3 i32x4* as_lds_v4(__bf16* p) {
    return (AS3 i32x4*)p;
}
#endif

static __device__ inline __bf16 f2bf(float f) {
    unsigned u = __builtin_bit_cast(unsigned, f);
    unsigned r = (u + 0x7FFFu + ((u >> 16) & 1u)) >> 16;
    unsigned short s = (unsigned short)r;
    return __builtin_bit_cast(__bf16, s);
}

static __device__ inline v8f_t vzero8() {
    v8f_t z;
#pragma unroll
    for (int i = 0; i < 8; ++i) z[i] = 0.0f;
    return z;
}

static __device__ inline v8f_t wmma_bf16(bfvec16 a, bfvec16 b, v8f_t c) {
    // D = A(16x32 bf16) * B(32x16 bf16) + C(16x16 f32)
    return __builtin_amdgcn_wmma_f32_16x16x32_bf16(
        /*neg_a=*/false, a, /*neg_b=*/false, b,
        /*c_mod=*/(short)0, c, /*reuse_a=*/false, /*reuse_b=*/false);
}

// ---------------------------------------------------------------------------
// Elementwise conversion kernels
// ---------------------------------------------------------------------------
__global__ void cvt_f32_bf16(const float* __restrict__ in,
                             __bf16* __restrict__ out, int n) {
    int i = blockIdx.x * 256 + threadIdx.x;
    if (i < n) out[i] = f2bf(in[i]);
}

// out[c*R + r] = bf16(in[r*C + c])   (transpose + convert)
__global__ void cvt_transpose(const float* __restrict__ in,
                              __bf16* __restrict__ out, int R, int C) {
    int i = blockIdx.x * 256 + threadIdx.x;
    if (i >= R * C) return;
    int r = i / C, c = i - r * C;
    out[(size_t)c * R + r] = f2bf(in[i]);
}

// RoPE on fp32 [S][nh*128] -> bf16 same layout.
__global__ void rope_cvt(const float* __restrict__ in,
                         __bf16* __restrict__ out, int nheads) {
    int i = blockIdx.x * 256 + threadIdx.x;
    int total = SEQ * nheads * 64;
    if (i >= total) return;
    int d = i & 63;
    int h = (i >> 6) % nheads;
    int s = i / (64 * nheads);
    size_t base = (size_t)s * (nheads * HEADDIM) + h * HEADDIM;
    float a = in[base + d];
    float b = in[base + 64 + d];
    // inv_freq = 10000^(-2d/128) ; ln(10000) = 9.210340371976184
    float inv = __expf(-((float)(2 * d) * (1.0f / 128.0f)) * 9.210340371976184f);
    float ang = (float)s * inv;
    float cs = cosf(ang), sn = sinf(ang);
    out[base + d]      = f2bf(a * cs - b * sn);
    out[base + 64 + d] = f2bf(b * cs + a * sn);
}

// ---------------------------------------------------------------------------
// bf16 GEMM: C[M,N] f32 = A[M,K] bf16 (row major) * Bt[N,K] bf16 (B transposed)
// Block: 256 thr (8 waves), block tile 256(M) x 128(N); wave = 64x64; K-step 32.
// 16 WMMAs per K-step per wave against 16 fragment ds_loads.
// ---------------------------------------------------------------------------
#define LDS_STRIDE 40   // 32 + 8 pad (bf16 elements)

__global__ __launch_bounds__(256) void gemm_bf16(
        const __bf16* __restrict__ A, const __bf16* __restrict__ Bt,
        float* __restrict__ C, int M, int N, int K) {
    __shared__ __align__(16) __bf16 As[256 * LDS_STRIDE];
    __shared__ __align__(16) __bf16 Bs[128 * LDS_STRIDE];

    const int t    = threadIdx.x;
    const int lane = t & 31;
    const int wid  = t >> 5;
    const int wm   = wid >> 1;      // 0..3 -> M offset wm*64
    const int wn   = wid & 1;       // 0..1 -> N offset wn*64
    const int c16  = lane & 15;
    const int half = lane >> 4;

    const int bm = blockIdx.y * 256;
    const int bn = blockIdx.x * 128;

    v8f_t acc[4][4];
#pragma unroll
    for (int mt = 0; mt < 4; ++mt)
#pragma unroll
        for (int nt = 0; nt < 4; ++nt) acc[mt][nt] = vzero8();

    // staging: thread handles A rows (r, r+128) and B row r, 16 bf16 each
    const int r  = t >> 1;            // 0..127
    const int cc = (t & 1) * 16;      // 0 or 16
    const __bf16* aSrc0 = A  + (size_t)(bm + r) * K + cc;
    const __bf16* aSrc1 = A  + (size_t)(bm + r + 128) * K + cc;
    const __bf16* bSrc  = Bt + (size_t)(bn + r) * K + cc;
    __bf16* aDst0 = &As[r * LDS_STRIDE + cc];
    __bf16* aDst1 = &As[(r + 128) * LDS_STRIDE + cc];
    __bf16* bDst  = &Bs[r * LDS_STRIDE + cc];

    for (int k0 = 0; k0 < K; k0 += 32) {
#if USE_ASYNC_LDS
        // direct global -> LDS async copies (ASYNCcnt), no VGPR bounce
        __builtin_amdgcn_global_load_async_to_lds_b128(
            as_global_v4(aSrc0 + k0),     as_lds_v4(aDst0),     0, 0);
        __builtin_amdgcn_global_load_async_to_lds_b128(
            as_global_v4(aSrc0 + k0 + 8), as_lds_v4(aDst0 + 8), 0, 0);
        __builtin_amdgcn_global_load_async_to_lds_b128(
            as_global_v4(aSrc1 + k0),     as_lds_v4(aDst1),     0, 0);
        __builtin_amdgcn_global_load_async_to_lds_b128(
            as_global_v4(aSrc1 + k0 + 8), as_lds_v4(aDst1 + 8), 0, 0);
        __builtin_amdgcn_global_load_async_to_lds_b128(
            as_global_v4(bSrc + k0),      as_lds_v4(bDst),      0, 0);
        __builtin_amdgcn_global_load_async_to_lds_b128(
            as_global_v4(bSrc + k0 + 8),  as_lds_v4(bDst + 8),  0, 0);
        __builtin_amdgcn_s_wait_asynccnt(0);
#else
        ((u32x4*)aDst0)[0] = ((const u32x4*)(aSrc0 + k0))[0];
        ((u32x4*)aDst0)[1] = ((const u32x4*)(aSrc0 + k0))[1];
        ((u32x4*)aDst1)[0] = ((const u32x4*)(aSrc1 + k0))[0];
        ((u32x4*)aDst1)[1] = ((const u32x4*)(aSrc1 + k0))[1];
        ((u32x4*)bDst)[0]  = ((const u32x4*)(bSrc + k0))[0];
        ((u32x4*)bDst)[1]  = ((const u32x4*)(bSrc + k0))[1];
#endif
        __syncthreads();

        FragBf afr[4], bfr[4];
#pragma unroll
        for (int mt = 0; mt < 4; ++mt) {
            int base = (wm * 64 + mt * 16 + c16) * LDS_STRIDE;
            afr[mt].u[0] = *(const u32x4*)&As[base + half * 8];
            afr[mt].u[1] = *(const u32x4*)&As[base + 16 + half * 8];
        }
#pragma unroll
        for (int nt = 0; nt < 4; ++nt) {
            int base = (wn * 64 + nt * 16 + c16) * LDS_STRIDE;
            bfr[nt].u[0] = *(const u32x4*)&Bs[base + half * 8];
            bfr[nt].u[1] = *(const u32x4*)&Bs[base + 16 + half * 8];
        }
#pragma unroll
        for (int mt = 0; mt < 4; ++mt)
#pragma unroll
            for (int nt = 0; nt < 4; ++nt)
                acc[mt][nt] = wmma_bf16(afr[mt].v, bfr[nt].v, acc[mt][nt]);
        __syncthreads();
    }

#pragma unroll
    for (int mt = 0; mt < 4; ++mt) {
        int row0 = bm + wm * 64 + mt * 16 + half * 8;
#pragma unroll
        for (int nt = 0; nt < 4; ++nt) {
            int col = bn + wn * 64 + nt * 16 + c16;
#pragma unroll
            for (int i = 0; i < 8; ++i)
                C[(size_t)(row0 + i) * N + col] = acc[mt][nt][i];
        }
    }
}

// ---------------------------------------------------------------------------
// Flash attention: one wave per (16-query block, head). Online softmax.
// qb: bf16 [S][2048] (post-RoPE), kb: bf16 [S][512] (post-RoPE),
// vT: bf16 [512][S] (transposed), of: f32 [S][2048]
// ---------------------------------------------------------------------------
__global__ __launch_bounds__(32) void flash_attn(
        const __bf16* __restrict__ qb, const __bf16* __restrict__ kb,
        const __bf16* __restrict__ vT, float* __restrict__ of) {
    __shared__ __align__(16) __bf16 pLds[16 * 32];

    const int lane = threadIdx.x & 31;
    const int c16  = lane & 15;
    const int half = lane >> 4;
    const int qblk = blockIdx.x;          // 0..255
    const int h    = blockIdx.y;          // 0..15
    const int kvh  = h >> 2;              // GQA: 4 query heads per KV head
    const float scale = 0.08838834764831845f;   // 1/sqrt(128)

    // Resident Q fragments: 16 rows x 128 dims = 4 A-frags (K-chunks of 32)
    FragBf qf[4];
    const __bf16* qptr = qb + (size_t)(qblk * 16 + c16) * DMODEL + h * HEADDIM;
#pragma unroll
    for (int dc = 0; dc < 4; ++dc) {
        qf[dc].u[0] = *(const u32x4*)(qptr + dc * 32 + half * 8);
        qf[dc].u[1] = *(const u32x4*)(qptr + dc * 32 + 16 + half * 8);
    }

    v8f_t o[8];
#pragma unroll
    for (int nt = 0; nt < 8; ++nt) o[nt] = vzero8();
    float m[8], l[8];
#pragma unroll
    for (int i = 0; i < 8; ++i) { m[i] = -1e30f; l[i] = 0.0f; }

    const int nkb = (qblk >> 1) + 1;      // causal: key blocks of 32
    for (int kbk = 0; kbk < nkb; ++kbk) {
        // prefetch next key/value tiles (speculative, bounded)
        if (kbk + 1 < nkb) {
            __builtin_prefetch(kb + (size_t)((kbk + 1) * 32 + c16) * KVDIM
                                  + kvh * HEADDIM, 0, 3);
            __builtin_prefetch(vT + (size_t)(kvh * HEADDIM + c16) * SEQ
                                  + (kbk + 1) * 32, 0, 3);
        }

        // ---- S = Q * K^T (two 16-key tiles) ----
        v8f_t s0 = vzero8(), s1 = vzero8();
#pragma unroll
        for (int dc = 0; dc < 4; ++dc) {
            FragBf k0f, k1f;
            const __bf16* kp0 = kb + (size_t)(kbk * 32 + c16) * KVDIM
                                   + kvh * HEADDIM + dc * 32;
            const __bf16* kp1 = kb + (size_t)(kbk * 32 + 16 + c16) * KVDIM
                                   + kvh * HEADDIM + dc * 32;
            k0f.u[0] = *(const u32x4*)(kp0 + half * 8); k0f.u[1] = *(const u32x4*)(kp0 + 16 + half * 8);
            k1f.u[0] = *(const u32x4*)(kp1 + half * 8); k1f.u[1] = *(const u32x4*)(kp1 + 16 + half * 8);
            s0 = wmma_bf16(qf[dc].v, k0f.v, s0);
            s1 = wmma_bf16(qf[dc].v, k1f.v, s1);
        }

        // ---- scale + causal mask (only diagonal block needs masking) ----
        const bool lastb = (kbk == nkb - 1);
#pragma unroll
        for (int i = 0; i < 8; ++i) { s0[i] *= scale; s1[i] *= scale; }
        if (lastb) {
            int key0 = kbk * 32 + c16;
            int key1 = key0 + 16;
#pragma unroll
            for (int i = 0; i < 8; ++i) {
                int qi = qblk * 16 + half * 8 + i;
                if (key0 > qi) s0[i] = -1e30f;
                if (key1 > qi) s1[i] = -1e30f;
            }
        }

        // ---- online softmax update (row = half*8+i, cols across lanes) ----
        float alpha[8];
#pragma unroll
        for (int i = 0; i < 8; ++i) {
            float rm = fmaxf(s0[i], s1[i]);
            rm = fmaxf(rm, __shfl_xor(rm, 1, 16));
            rm = fmaxf(rm, __shfl_xor(rm, 2, 16));
            rm = fmaxf(rm, __shfl_xor(rm, 4, 16));
            rm = fmaxf(rm, __shfl_xor(rm, 8, 16));
            float mn = fmaxf(m[i], rm);
            float p0 = __expf(s0[i] - mn);
            float p1 = __expf(s1[i] - mn);
            s0[i] = p0; s1[i] = p1;
            float rs = p0 + p1;
            rs += __shfl_xor(rs, 1, 16);
            rs += __shfl_xor(rs, 2, 16);
            rs += __shfl_xor(rs, 4, 16);
            rs += __shfl_xor(rs, 8, 16);
            alpha[i] = __expf(m[i] - mn);
            l[i] = l[i] * alpha[i] + rs;
            m[i] = mn;
        }
#pragma unroll
        for (int nt = 0; nt < 8; ++nt)
#pragma unroll
            for (int i = 0; i < 8; ++i) o[nt][i] *= alpha[i];

        // ---- transpose P from C-layout to A-layout through LDS ----
#pragma unroll
        for (int i = 0; i < 8; ++i) {
            int row = half * 8 + i;
            pLds[row * 32 + c16]      = f2bf(s0[i]);
            pLds[row * 32 + 16 + c16] = f2bf(s1[i]);
        }
        __syncthreads();
        FragBf pf;
        pf.u[0] = *(const u32x4*)&pLds[c16 * 32 + half * 8];
        pf.u[1] = *(const u32x4*)&pLds[c16 * 32 + 16 + half * 8];
        __syncthreads();

        // ---- O += P * V  (V^T layout gives contiguous key runs) ----
#pragma unroll
        for (int nt = 0; nt < 8; ++nt) {
            FragBf vf;
            const __bf16* vp = vT + (size_t)(kvh * HEADDIM + nt * 16 + c16) * SEQ
                                  + kbk * 32;
            vf.u[0] = *(const u32x4*)(vp + half * 8);
            vf.u[1] = *(const u32x4*)(vp + 16 + half * 8);
            o[nt] = wmma_bf16(pf.v, vf.v, o[nt]);
        }
    }

    // ---- normalize and write fp32 [s][h*128 + d] ----
    float linv[8];
#pragma unroll
    for (int i = 0; i < 8; ++i) linv[i] = 1.0f / l[i];
    float* op = of + (size_t)(qblk * 16) * DMODEL + h * HEADDIM;
#pragma unroll
    for (int nt = 0; nt < 8; ++nt) {
#pragma unroll
        for (int i = 0; i < 8; ++i) {
            int row = half * 8 + i;
            op[(size_t)row * DMODEL + nt * 16 + c16] = o[nt][i] * linv[i];
        }
    }
}

// ---------------------------------------------------------------------------
// Host-side orchestration
// ---------------------------------------------------------------------------
extern "C" void kernel_launch(void* const* d_in, const int* in_sizes, int n_in,
                              void* d_out, int out_size, void* d_ws, size_t ws_size,
                              hipStream_t stream) {
    const float* x  = (const float*)d_in[0];
    const float* wq = (const float*)d_in[1];
    const float* wk = (const float*)d_in[2];
    const float* wv = (const float*)d_in[3];
    const float* wo = (const float*)d_in[4];
    float* out = (float*)d_out;

    char* ws = (char*)d_ws;
    size_t off = 0;
    auto take = [&](size_t bytes) -> char* {
        char* p = ws + off;
        off += (bytes + 255) & ~(size_t)255;
        return p;
    };

    __bf16* xb  = (__bf16*)take((size_t)SEQ * DMODEL * 2);
    __bf16* wqT = (__bf16*)take((size_t)DMODEL * DMODEL * 2);
    __bf16* wkT = (__bf16*)take((size_t)DMODEL * KVDIM * 2);
    __bf16* wvT = (__bf16*)take((size_t)DMODEL * KVDIM * 2);
    __bf16* woT = (__bf16*)take((size_t)DMODEL * DMODEL * 2);
    float*  qf  = (float*)take((size_t)SEQ * DMODEL * 4);
    float*  kf  = (float*)take((size_t)SEQ * KVDIM * 4);
    float*  vf  = (float*)take((size_t)SEQ * KVDIM * 4);
    __bf16* qbB = (__bf16*)take((size_t)SEQ * DMODEL * 2);
    __bf16* kbB = (__bf16*)take((size_t)SEQ * KVDIM * 2);
    __bf16* vT  = (__bf16*)take((size_t)KVDIM * SEQ * 2);
    float*  of  = (float*)take((size_t)SEQ * DMODEL * 4);
    __bf16* ob  = (__bf16*)take((size_t)SEQ * DMODEL * 2);

    // 1. convert activations / transpose+convert weights to bf16
    cvt_f32_bf16<<<(SEQ * DMODEL) / 256, 256, 0, stream>>>(x, xb, SEQ * DMODEL);
    cvt_transpose<<<(DMODEL * DMODEL) / 256, 256, 0, stream>>>(wq, wqT, DMODEL, DMODEL);
    cvt_transpose<<<(DMODEL * KVDIM) / 256, 256, 0, stream>>>(wk, wkT, DMODEL, KVDIM);
    cvt_transpose<<<(DMODEL * KVDIM) / 256, 256, 0, stream>>>(wv, wvT, DMODEL, KVDIM);
    cvt_transpose<<<(DMODEL * DMODEL) / 256, 256, 0, stream>>>(wo, woT, DMODEL, DMODEL);

    // 2. QKV projections (bf16 WMMA, f32 accumulate)
    gemm_bf16<<<dim3(DMODEL / 128, SEQ / 256), 256, 0, stream>>>(xb, wqT, qf, SEQ, DMODEL, DMODEL);
    gemm_bf16<<<dim3(KVDIM / 128, SEQ / 256), 256, 0, stream>>>(xb, wkT, kf, SEQ, KVDIM, DMODEL);
    gemm_bf16<<<dim3(KVDIM / 128, SEQ / 256), 256, 0, stream>>>(xb, wvT, vf, SEQ, KVDIM, DMODEL);

    // 3. RoPE + bf16 downconvert for Q and K; transpose+convert V
    rope_cvt<<<(SEQ * NHEADS * 64) / 256, 256, 0, stream>>>(qf, qbB, NHEADS);
    rope_cvt<<<(SEQ * NKVH * 64) / 256, 256, 0, stream>>>(kf, kbB, NKVH);
    cvt_transpose<<<(SEQ * KVDIM) / 256, 256, 0, stream>>>(vf, vT, SEQ, KVDIM);

    // 4. causal flash attention (one wave per 16 query rows per head)
    flash_attn<<<dim3(SEQ / 16, NHEADS), 32, 0, stream>>>(qbB, kbB, vT, of);

    // 5. output projection
    cvt_f32_bf16<<<(SEQ * DMODEL) / 256, 256, 0, stream>>>(of, ob, SEQ * DMODEL);
    gemm_bf16<<<dim3(DMODEL / 128, SEQ / 256), 256, 0, stream>>>(ob, woT, out, SEQ, DMODEL, DMODEL);
}
